// BiLSTM_45234595561814
// MI455X (gfx1250) — matrix operand order repaired and checked by reading, not compiled, not run on hardware
//
#include <hip/hip_runtime.h>
#include <hip/hip_bf16.h>
#include <math.h>

// ---------------------------------------------------------------------------
// BiLSTM (2 layers, H=128) + dense head for MI455X (gfx1250).
//  - All matmuls via v_wmma_f32_16x16x32_f16 (f16 operands, f32 accum).
//  - xg (input projection) produced in WMMA C-fragment order so the
//    sequential recurrence initializes accumulators with ONE b128 load
//    per lane per tile (instead of 8 strided u16 loads).
//  - Recurrence: persistent workgroup, Whh resident in LDS (128KB),
//    h/gates staged through LDS, cell state c in f32 registers.
// ---------------------------------------------------------------------------

#define B_   64
#define T_   512
#define D_   64
#define H_   128
#define G_   512   // 4*H
#define O2H_ 256   // 2*H
#define NT_  32    // G/16 n-tiles per step

typedef __attribute__((ext_vector_type(16))) _Float16     v16h;
typedef __attribute__((ext_vector_type(8)))  float        v8f;
typedef __attribute__((ext_vector_type(4)))  unsigned int u32x4;

union FragH  { v16h h; u32x4 q[2]; };
union Frag8H { _Float16 h[8]; u32x4 q; };

// A fragment: 16x32 f16 (MxK); rows m stride ldrow, K contiguous.
// Lane l: m = l&15, koff = (l>>4)*8 ; V0-3 = K[koff..+7], V4-7 = K[16+koff..+7]
__device__ __forceinline__ v16h frag_a(const _Float16* row0, int ldrow, int k0, int lane) {
  FragH f;
  const int m = lane & 15, koff = (lane >> 4) * 8;
  const _Float16* p = row0 + (size_t)m * ldrow + k0 + koff;
  f.q[0] = *(const u32x4*)(p);
  f.q[1] = *(const u32x4*)(p + 16);
  return f.h;
}
// B fragment: 32x16 f16 (KxN) from weight stored [N][K] row-major.
// Lane l: n = l&15, kb = (l>>4)*8 ; V0-3 = K[kb..+7], V4-7 = K[16+kb..+7]
__device__ __forceinline__ v16h frag_b(const _Float16* wrow0, int ld, int k0, int lane) {
  FragH f;
  const int n = lane & 15, kb = (lane >> 4) * 8;
  const _Float16* p = wrow0 + (size_t)n * ld + k0 + kb;
  f.q[0] = *(const u32x4*)(p);
  f.q[1] = *(const u32x4*)(p + 16);
  return f.h;
}
__device__ __forceinline__ v8f wmma16(v16h a, v16h b, v8f c) {
  return __builtin_amdgcn_wmma_f32_16x16x32_f16(false, a, false, b, (short)0, c, false, false);
}
__device__ __forceinline__ float sigm_(float x) { return 1.0f / (1.0f + __expf(-x)); }

// --------------------------- small prep kernels ----------------------------
__global__ void k_f32_to_f16(const float* __restrict__ s, _Float16* __restrict__ d, int n) {
  int i = blockIdx.x * blockDim.x + threadIdx.x;
  if (i < n) d[i] = (_Float16)s[i];
}
__global__ void k_bias_sum(const float* __restrict__ a, const float* __restrict__ b,
                           float* __restrict__ o, int n) {
  int i = blockIdx.x * blockDim.x + threadIdx.x;
  if (i < n) o[i] = a[i] + b[i];
}

// ------------------- xg projection GEMM (fragment output) ------------------
// For one direction: xg_frag[bs][t][nt][lane][r] (f16) =
//   A[(bs*16+m)*T + t][:] . W[nt*16+n][:] + bias, tile rows = 16 batch rows.
// Linear tile id = ((bs*T + t)*NT_ + nt); store at tile*256 + lane*8.
__global__ void k_gemm_xg(const _Float16* __restrict__ A, int lda,   // [B*T][K]
                          const _Float16* __restrict__ W,            // [G][K]
                          const float* __restrict__ bias,            // [G]
                          _Float16* __restrict__ xgf,                // frag layout
                          int K) {
  const int lane = threadIdx.x & 31;
  const int wv   = threadIdx.x >> 5;
  const int tile = blockIdx.x * (blockDim.x >> 5) + wv;
  const int total = (B_ / 16) * T_ * NT_;
  if (tile >= total) return;
  const int bs = tile / (T_ * NT_);
  const int r1 = tile % (T_ * NT_);
  const int t  = r1 / NT_;
  const int nt = r1 % NT_;

  const float bv = bias[nt * 16 + (lane & 15)];
  v8f c;
#pragma unroll
  for (int r = 0; r < 8; ++r) c[r] = bv;

  const _Float16* Arow = A + ((size_t)(bs * 16) * T_ + t) * lda;  // row stride T*lda
  const _Float16* Wrow = W + (size_t)nt * 16 * K;
  for (int k0 = 0; k0 < K; k0 += 32) {
    v16h a = frag_a(Arow, T_ * lda, k0, lane);
    v16h b = frag_b(Wrow, K, k0, lane);
    c = wmma16(a, b, c);
  }
  Frag8H o;
#pragma unroll
  for (int r = 0; r < 8; ++r) o.h[r] = (_Float16)c[r];
  *(u32x4*)(xgf + (size_t)tile * 256 + lane * 8) = o.q;
}

// --------------------------- generic WMMA GEMM -----------------------------
// C[M][N] = A[M][K] * W[N][K]^T + bias[N], one 16x16 tile per wave.
// EPI: 1 = relu -> f16 ; 2 = tanh -> f32
template <int EPI>
__global__ void k_gemm(const _Float16* __restrict__ A, int lda,
                       const _Float16* __restrict__ W, int ldw,
                       const float* __restrict__ bias,
                       void* __restrict__ Cout, int ldc,
                       int Mtiles, int Ntiles, int K) {
  const int lane = threadIdx.x & 31;
  const int wv   = threadIdx.x >> 5;
  const int tile = blockIdx.x * (blockDim.x >> 5) + wv;
  if (tile >= Mtiles * Ntiles) return;
  const int mt = tile / Ntiles, nt = tile % Ntiles;

  const float bv = bias[nt * 16 + (lane & 15)];
  v8f c;
#pragma unroll
  for (int r = 0; r < 8; ++r) c[r] = bv;

  const _Float16* Arow = A + (size_t)mt * 16 * lda;
  const _Float16* Wrow = W + (size_t)nt * 16 * ldw;
  for (int k0 = 0; k0 < K; k0 += 32) {
    v16h a = frag_a(Arow, lda, k0, lane);
    v16h b = frag_b(Wrow, ldw, k0, lane);
    c = wmma16(a, b, c);
  }

  const int col   = nt * 16 + (lane & 15);
  const int rbase = mt * 16 + ((lane >> 4) << 3);
  if (EPI == 2) {
    float* C = (float*)Cout;
#pragma unroll
    for (int r = 0; r < 8; ++r) C[(size_t)(rbase + r) * ldc + col] = tanhf(c[r]);
  } else {
    _Float16* C = (_Float16*)Cout;
#pragma unroll
    for (int r = 0; r < 8; ++r) {
      float v = c[r];
      if (EPI == 1) v = v > 0.f ? v : 0.f;
      C[(size_t)(rbase + r) * ldc + col] = (_Float16)v;
    }
  }
}

// --------------------------- LSTM recurrence -------------------------------
// Grid (B/16, 2): blockIdx.x = batch slab of 16 rows, blockIdx.y = direction.
// Per step: g[16][512] = xg_frag[:,tt,:] + h @ Whh^T (WMMA, Whh in LDS), gates.
__global__ void __launch_bounds__(512)
k_lstm(const _Float16* __restrict__ xg,    // [2][bs][T][nt][256] fragment layout
       const _Float16* __restrict__ WhhF,  // [G][H] f16
       const _Float16* __restrict__ WhhB,  // [G][H] f16
       _Float16* __restrict__ out) {       // [B][T][2H] f16; dir -> col offset
  extern __shared__ char smem[];
  _Float16* Whs  = (_Float16*)smem;                                    // 128 KB
  _Float16* hbuf = (_Float16*)(smem + (size_t)G_ * H_ * 2);            //   4 KB
  float*    gbuf = (float*)(smem + (size_t)G_ * H_ * 2 + 16 * H_ * 2); //  32 KB

  const int tid  = threadIdx.x;
  const int lane = tid & 31, wv = tid >> 5;
  const int dir  = blockIdx.y;
  const int bs   = blockIdx.x;
  const int b0   = bs * 16;
  const _Float16* Whh = dir ? WhhB : WhhF;
  const _Float16* xgd = xg + (size_t)dir * B_ * T_ * G_;  // per-dir frag block

  { // stage Whh into LDS (8192 x 16B)
    const u32x4* src = (const u32x4*)Whh;
    u32x4*       dst = (u32x4*)Whs;
    for (int i = tid; i < (G_ * H_ / 8); i += 512) dst[i] = src[i];
  }
  for (int i = tid; i < 16 * H_; i += 512) hbuf[i] = (_Float16)0.f;
  float cc[4] = {0.f, 0.f, 0.f, 0.f};
  __syncthreads();

  for (int t = 0; t < T_; ++t) {
    const int tt = dir ? (T_ - 1 - t) : t;
    // Phase A: 16 waves x 2 N-tiles = 32 tiles covering [16][512]
#pragma unroll
    for (int nn = 0; nn < 2; ++nn) {
      const int nt  = wv + nn * 16;
      const int col = nt * 16 + (lane & 15);
      const int rb  = (lane >> 4) << 3;
      // accumulator init: one 16B load in fragment order
      Frag8H xi;
      xi.q = *(const u32x4*)(xgd + ((((size_t)bs * T_ + tt) * NT_ + nt) * 256) + lane * 8);
      v8f acc;
#pragma unroll
      for (int r = 0; r < 8; ++r) acc[r] = (float)xi.h[r];
      // prefetch next step's tile while this step computes
      if (t + 1 < T_) {
        const int tn = dir ? (tt - 1) : (tt + 1);
        __builtin_prefetch(xgd + ((((size_t)bs * T_ + tn) * NT_ + nt) * 256) + lane * 8, 0, 1);
      }
      const _Float16* Wrow = Whs + (size_t)nt * 16 * H_;
#pragma unroll
      for (int k0 = 0; k0 < H_; k0 += 32) {
        v16h a = frag_a(hbuf, H_, k0, lane);
        v16h b = frag_b(Wrow, H_, k0, lane);
        acc = wmma16(a, b, acc);
      }
#pragma unroll
      for (int r = 0; r < 8; ++r) gbuf[(rb + r) * G_ + col] = acc[r];
    }
    __syncthreads();
    // Phase B: gate nonlinearities + state update (thread owns 4 fixed (m,j))
#pragma unroll
    for (int k = 0; k < 4; ++k) {
      const int e = tid + k * 512;
      const int m = e >> 7, j = e & 127;
      const float gi = sigm_(gbuf[m * G_ + j]);
      const float gf = sigm_(gbuf[m * G_ + H_ + j]);
      const float gg = tanhf(gbuf[m * G_ + 2 * H_ + j]);
      const float go = sigm_(gbuf[m * G_ + 3 * H_ + j]);
      cc[k] = gf * cc[k] + gi * gg;
      const float h = go * tanhf(cc[k]);
      hbuf[m * H_ + j] = (_Float16)h;
      out[((size_t)(b0 + m) * T_ + tt) * O2H_ + dir * H_ + j] = (_Float16)h;
    }
    __syncthreads();
  }
}

// ---------------------------------------------------------------------------
extern "C" void kernel_launch(void* const* d_in, const int* in_sizes, int n_in,
                              void* d_out, int out_size, void* d_ws, size_t ws_size,
                              hipStream_t stream) {
  (void)in_sizes; (void)n_in; (void)out_size; (void)ws_size;
  const float* x     = (const float*)d_in[0];
  const float* Wih00 = (const float*)d_in[1];
  const float* Whh00 = (const float*)d_in[2];
  const float* bih00 = (const float*)d_in[3];
  const float* bhh00 = (const float*)d_in[4];
  const float* Wih01 = (const float*)d_in[5];
  const float* Whh01 = (const float*)d_in[6];
  const float* bih01 = (const float*)d_in[7];
  const float* bhh01 = (const float*)d_in[8];
  const float* Wih10 = (const float*)d_in[9];
  const float* Whh10 = (const float*)d_in[10];
  const float* bih10 = (const float*)d_in[11];
  const float* bhh10 = (const float*)d_in[12];
  const float* Wih11 = (const float*)d_in[13];
  const float* Whh11 = (const float*)d_in[14];
  const float* bih11 = (const float*)d_in[15];
  const float* bhh11 = (const float*)d_in[16];
  const float* W1    = (const float*)d_in[17];
  const float* b1    = (const float*)d_in[18];
  const float* W2    = (const float*)d_in[19];
  const float* b2    = (const float*)d_in[20];

  // ---- workspace carve (~90 MB) ----
  char* w = (char*)d_ws;
  auto take = [&](size_t bytes) -> char* {
    char* p = w; w += (bytes + 255) & ~(size_t)255; return p;
  };
  _Float16* xh     = (_Float16*)take((size_t)B_ * T_ * D_ * 2);
  _Float16* Wih00h = (_Float16*)take((size_t)G_ * D_ * 2);
  _Float16* Wih01h = (_Float16*)take((size_t)G_ * D_ * 2);
  _Float16* Wih10h = (_Float16*)take((size_t)G_ * O2H_ * 2);
  _Float16* Wih11h = (_Float16*)take((size_t)G_ * O2H_ * 2);
  _Float16* Whh00h = (_Float16*)take((size_t)G_ * H_ * 2);
  _Float16* Whh01h = (_Float16*)take((size_t)G_ * H_ * 2);
  _Float16* Whh10h = (_Float16*)take((size_t)G_ * H_ * 2);
  _Float16* Whh11h = (_Float16*)take((size_t)G_ * H_ * 2);
  _Float16* W1h    = (_Float16*)take((size_t)256 * O2H_ * 2);
  _Float16* W2h    = (_Float16*)take((size_t)D_ * 256 * 2);
  float* bs00 = (float*)take(G_ * 4);
  float* bs01 = (float*)take(G_ * 4);
  float* bs10 = (float*)take(G_ * 4);
  float* bs11 = (float*)take(G_ * 4);
  _Float16* xgbuf = (_Float16*)take((size_t)2 * B_ * T_ * G_ * 2);  // 67 MB (frag layout)
  _Float16* out0  = (_Float16*)take((size_t)B_ * T_ * O2H_ * 2);    // 16 MB
  _Float16* out1  = out0;   // layer1 recurrence only reads xg; safe alias
  _Float16* mid   = xgbuf;  // xg dead after layer1 recurrence; safe alias

  // ---- f16 weight/input staging ----
  auto cvt = [&](const float* s, _Float16* d, int n) {
    k_f32_to_f16<<<(n + 255) / 256, 256, 0, stream>>>(s, d, n);
  };
  cvt(x, xh, B_ * T_ * D_);
  cvt(Wih00, Wih00h, G_ * D_);   cvt(Wih01, Wih01h, G_ * D_);
  cvt(Wih10, Wih10h, G_ * O2H_); cvt(Wih11, Wih11h, G_ * O2H_);
  cvt(Whh00, Whh00h, G_ * H_);   cvt(Whh01, Whh01h, G_ * H_);
  cvt(Whh10, Whh10h, G_ * H_);   cvt(Whh11, Whh11h, G_ * H_);
  cvt(W1, W1h, 256 * O2H_);      cvt(W2, W2h, D_ * 256);
  k_bias_sum<<<2, 256, 0, stream>>>(bih00, bhh00, bs00, G_);
  k_bias_sum<<<2, 256, 0, stream>>>(bih01, bhh01, bs01, G_);
  k_bias_sum<<<2, 256, 0, stream>>>(bih10, bhh10, bs10, G_);
  k_bias_sum<<<2, 256, 0, stream>>>(bih11, bhh11, bs11, G_);

  const int    MT      = (B_ * T_) / 16;  // 2048 M-tiles
  const size_t dstride = (size_t)B_ * T_ * G_;
  const size_t smemrec = (size_t)G_ * H_ * 2 + 16 * H_ * 2 + 16 * G_ * 4; // 164 KB
  (void)hipFuncSetAttribute((const void*)k_lstm,
                            hipFuncAttributeMaxDynamicSharedMemorySize, (int)smemrec);

  const int xg_tiles  = (B_ / 16) * T_ * NT_;  // 65536 tiles per direction
  const int xg_blocks = (xg_tiles + 7) / 8;

  // ---- layer 0 ----
  k_gemm_xg<<<xg_blocks, 256, 0, stream>>>(xh, D_, Wih00h, bs00, xgbuf, D_);
  k_gemm_xg<<<xg_blocks, 256, 0, stream>>>(xh, D_, Wih01h, bs01, xgbuf + dstride, D_);
  k_lstm<<<dim3(B_ / 16, 2), 512, smemrec, stream>>>(xgbuf, Whh00h, Whh01h, out0);

  // ---- layer 1 ----
  k_gemm_xg<<<xg_blocks, 256, 0, stream>>>(out0, O2H_, Wih10h, bs10, xgbuf, O2H_);
  k_gemm_xg<<<xg_blocks, 256, 0, stream>>>(out0, O2H_, Wih11h, bs11, xgbuf + dstride, O2H_);
  k_lstm<<<dim3(B_ / 16, 2), 512, smemrec, stream>>>(xgbuf, Whh10h, Whh11h, out1);

  // ---- dense head: relu(out1 @ W1^T + b1) -> tanh(mid @ W2^T + b2) ----
  int tiles1 = MT * (256 / 16);
  k_gemm<1><<<(tiles1 + 7) / 8, 256, 0, stream>>>(out1, O2H_, W1h, O2H_, b1,
                                                  mid, 256, MT, 256 / 16, O2H_);
  int tiles2 = MT * (D_ / 16);
  k_gemm<2><<<(tiles2 + 7) / 8, 256, 0, stream>>>(mid, 256, W2h, 256, b2,
                                                  d_out, D_, MT, D_ / 16, 256);
}
